// EfficientSelfAttention_7791070675638
// MI455X (gfx1250) — matrix operand order repaired and checked by reading, not compiled, not run on hardware
//
#include <hip/hip_runtime.h>
#include <hip/hip_bf16.h>
#include <hip/hip_fp16.h>

typedef __attribute__((ext_vector_type(16))) _Float16 v16h;
typedef __attribute__((ext_vector_type(8)))  _Float16 v8h;
typedef __attribute__((ext_vector_type(8)))  float    v8f;

// Problem constants (fixed by the reference)
#define BZ    8
#define NQ    4096
#define CC    256
#define NHEAD 8
#define DHEAD 32
#define MRED  256      // (64/4)*(64/4)
#define KCONV 4096     // C*R*R

// ---------------------------------------------------------------------------
// CDNA5 async copy to LDS (ASYNCcnt-tracked) — gfx1250 path
// ---------------------------------------------------------------------------
__device__ __forceinline__ void async_copy16(unsigned lds_byte_addr,
                                             const void* gaddr) {
  // GLOBAL_LOAD_ASYNC_TO_LDS_B128, GV mode (64-bit vaddr, saddr=off)
  asm volatile("global_load_async_to_lds_b128 %0, %1, off"
               :: "v"(lds_byte_addr), "v"(gaddr)
               : "memory");
}

__device__ __forceinline__ void wait_async0() {
#if __has_builtin(__builtin_amdgcn_s_wait_asynccnt)
  __builtin_amdgcn_s_wait_asynccnt(0);
#else
  asm volatile("s_wait_asynccnt 0x0" ::: "memory");
#endif
}

__device__ __forceinline__ unsigned lds_addr_of(const void* p) {
  // flat shared address: low 32 bits == wave-relative LDS byte address
  return (unsigned)(unsigned long long)p;
}

// ---------------------------------------------------------------------------
// WMMA fragment helpers (v_wmma_f32_16x16x32_f16, wave32; contiguous k = 16*hi + s in A and B)
// ---------------------------------------------------------------------------
__device__ __forceinline__ v16h frag_a_row(const _Float16* rowp, int k0, int hi) {
  v8h c0 = *(const v8h*)(rowp + k0 + (hi ? 16 : 0));
  v8h c1 = *(const v8h*)(rowp + k0 + (hi ? 24 : 8));
  v16h r;
#pragma unroll
  for (int i = 0; i < 8; ++i) { r[i] = c0[i]; r[8 + i] = c1[i]; }
  return r;
}

__device__ __forceinline__ v16h frag_b_col(const _Float16* colp, int k0, int hi) {
  return *(const v16h*)(colp + k0 + (hi ? 16 : 0));
}

__device__ __forceinline__ v8f wmma_f16(v16h a, v16h b, v8f c) {
  return __builtin_amdgcn_wmma_f32_16x16x32_f16(false, a, false, b, (short)0, c,
                                                false, false);
}

// ---------------------------------------------------------------------------
// Packing kernels
// ---------------------------------------------------------------------------
__global__ void f32_to_f16_kernel(const float* __restrict__ src,
                                  _Float16* __restrict__ dst, size_t n) {
  size_t i = (size_t)blockIdx.x * blockDim.x + threadIdx.x;
  size_t stride = (size_t)gridDim.x * blockDim.x;
  for (; i < n; i += stride) dst[i] = (_Float16)src[i];
}

// conv_w OIHW (256,256,4,4) -> wc (o, k) with k = (kh*4+kw)*256 + i
__global__ void pack_conv_w_kernel(const float* __restrict__ cw,
                                   _Float16* __restrict__ wc) {
  int idx = blockIdx.x * blockDim.x + threadIdx.x;
  if (idx >= CC * KCONV) return;
  int o  = idx >> 12;
  int k  = idx & 4095;
  int kk = k >> 8;
  int i  = k & 255;
  wc[idx] = (_Float16)cw[(o * 256 + i) * 16 + kk];
}

// ---------------------------------------------------------------------------
// Stage 1: strided 4x4 conv as implicit-im2col GEMM (2048 x 256, K=4096)
// One block per 16-row strip; patch chunk (16x256) async-staged per position.
// 8 waves x 2 column tiles.
// ---------------------------------------------------------------------------
__global__ __launch_bounds__(256)
void conv_gemm_kernel(const _Float16* __restrict__ xh,   // (B,4096,256) f16
                      const _Float16* __restrict__ wc,   // (256,4096)   f16
                      const float* __restrict__ conv_b,
                      float* __restrict__ x1_pre) {      // (2048,256)   f32
  __shared__ __align__(16) _Float16 atile[16 * 264];     // padded, 8.25 KB

  const int wave = threadIdx.x >> 5, lane = threadIdx.x & 31;
  const int hi = lane >> 4, l16 = lane & 15;
  const int strip = blockIdx.x;                          // 0..127
  const int tn0 = wave * 2;

  const _Float16* wrow0 = wc + (size_t)((tn0 + 0) * 16 + l16) * KCONV;
  const _Float16* wrow1 = wc + (size_t)((tn0 + 1) * 16 + l16) * KCONV;
  const unsigned t = threadIdx.x;
  const unsigned ldsA = lds_addr_of(&atile[0]);

  v8f acc0 = {}, acc1 = {};
#pragma unroll 1
  for (int kk = 0; kk < 16; ++kk) {
    const int kh = kk >> 2, kw = kk & 3;
#pragma unroll
    for (int i = 0; i < 2; ++i) {                        // 512 x 16B granules
      const unsigned g = t + i * 256;
      const int row = g >> 5, seg = g & 31;
      const int grow = strip * 16 + row;                 // b*256+m
      const int b = grow >> 8, m = grow & 255;
      const int tok = ((m >> 4) * 4 + kh) * 64 + ((m & 15) * 4 + kw);
      async_copy16(ldsA + row * 528 + seg * 16,
                   xh + ((size_t)(b * NQ + tok)) * CC + seg * 8);
    }
    wait_async0();
    __syncthreads();
    const _Float16* arow = &atile[l16 * 264];
#pragma unroll
    for (int k2 = 0; k2 < 8; ++k2) {
      v16h af = frag_a_row(arow, k2 * 32, hi);
      acc0 = wmma_f16(af, frag_b_col(wrow0, kk * 256 + k2 * 32, hi), acc0);
      acc1 = wmma_f16(af, frag_b_col(wrow1, kk * 256 + k2 * 32, hi), acc1);
    }
    __syncthreads();
  }
  const float b0 = conv_b[(tn0 + 0) * 16 + l16];
  const float b1 = conv_b[(tn0 + 1) * 16 + l16];
#pragma unroll
  for (int r = 0; r < 8; ++r) {
    int row = strip * 16 + hi * 8 + r;
    x1_pre[(size_t)row * CC + (tn0 + 0) * 16 + l16] = acc0[r] + b0;
    x1_pre[(size_t)row * CC + (tn0 + 1) * 16 + l16] = acc1[r] + b1;
  }
}

// ---------------------------------------------------------------------------
// Stage 2: LayerNorm over C=256, fp32 in -> f16 out. One wave per row.
// ---------------------------------------------------------------------------
__global__ __launch_bounds__(256)
void ln_kernel(const float* __restrict__ x1_pre,
               const float* __restrict__ ln_g, const float* __restrict__ ln_b,
               _Float16* __restrict__ x1h) {
  const int wave = threadIdx.x >> 5, lane = threadIdx.x & 31;
  const int row = blockIdx.x * 8 + wave;
  const float* rp = x1_pre + (size_t)row * CC;
  float v[8], s = 0.f, s2 = 0.f;
#pragma unroll
  for (int i = 0; i < 8; ++i) {
    v[i] = rp[lane + i * 32];
    s += v[i]; s2 += v[i] * v[i];
  }
#pragma unroll
  for (int off = 16; off > 0; off >>= 1) {
    s  += __shfl_xor(s,  off, 32);
    s2 += __shfl_xor(s2, off, 32);
  }
  const float mu   = s * (1.f / 256.f);
  const float var  = s2 * (1.f / 256.f) - mu * mu;
  const float rstd = rsqrtf(var + 1e-5f);
  _Float16* op = x1h + (size_t)row * CC;
#pragma unroll
  for (int i = 0; i < 8; ++i) {
    int c = lane + i * 32;
    op[c] = (_Float16)((v[i] - mu) * rstd * ln_g[c] + ln_b[c]);
  }
}

// ---------------------------------------------------------------------------
// Stage 3: KV projection (2048 x 512, K=256). One block per 16-row strip,
// A strip async-staged once; 8 waves x 4 column tiles (32 total).
// K -> (bh,m,d), V -> (bh,d,m) transposed.
// ---------------------------------------------------------------------------
__global__ __launch_bounds__(256)
void kv_gemm_kernel(const _Float16* __restrict__ x1h,    // (2048,256)
                    const _Float16* __restrict__ kvw,    // (512,256)
                    const float* __restrict__ kv_b,
                    _Float16* __restrict__ kbuf,         // (64,256,32)
                    _Float16* __restrict__ vtb) {        // (64,32,256)
  __shared__ __align__(16) _Float16 atile[16 * 264];

  const int wave = threadIdx.x >> 5, lane = threadIdx.x & 31;
  const int hi = lane >> 4, l16 = lane & 15;
  const int strip = blockIdx.x;                          // 0..127
  const unsigned t = threadIdx.x;
  const unsigned ldsA = lds_addr_of(&atile[0]);

#pragma unroll
  for (int i = 0; i < 2; ++i) {                          // 512 granules
    const unsigned g = t + i * 256;
    const int row = g >> 5, seg = g & 31;
    async_copy16(ldsA + row * 528 + seg * 16,
                 x1h + (size_t)(strip * 16 + row) * CC + seg * 8);
  }
  wait_async0();
  __syncthreads();

  const _Float16* arow = &atile[l16 * 264];
  const _Float16* wr[4];
  v8f acc[4] = {};
#pragma unroll
  for (int u = 0; u < 4; ++u)
    wr[u] = kvw + (size_t)((wave * 4 + u) * 16 + l16) * CC;

#pragma unroll
  for (int k0 = 0; k0 < CC; k0 += 32) {
    v16h af = frag_a_row(arow, k0, hi);
#pragma unroll
    for (int u = 0; u < 4; ++u)
      acc[u] = wmma_f16(af, frag_b_col(wr[u], k0, hi), acc[u]);
  }

#pragma unroll
  for (int u = 0; u < 4; ++u) {
    const int n = (wave * 4 + u) * 16 + l16;             // 0..511
    const float bias = kv_b[n];
    const int sKV = n >> 8, h = (n >> 5) & 7, dd = n & 31;
#pragma unroll
    for (int r = 0; r < 8; ++r) {
      int row = strip * 16 + hi * 8 + r;                 // b*256+m
      int b_ = row >> 8, m = row & 255;
      float val = acc[u][r] + bias;
      if (sKV == 0)
        kbuf[(((size_t)(b_ * NHEAD + h) * MRED) + m) * DHEAD + dd] = (_Float16)val;
      else
        vtb[(((size_t)(b_ * NHEAD + h) * DHEAD) + dd) * MRED + m] = (_Float16)val;
    }
  }
}

// ---------------------------------------------------------------------------
// Stage 4: Q projection (32768 x 256, K=256). One block (128 thr) per strip,
// A strip async-staged; 4 waves x 4 column tiles.
// ---------------------------------------------------------------------------
__global__ __launch_bounds__(128)
void q_gemm_kernel(const _Float16* __restrict__ xh,      // (32768,256)
                   const _Float16* __restrict__ qw,      // (256,256)
                   const float* __restrict__ q_b,
                   _Float16* __restrict__ qbuf) {        // (64,4096,32)
  __shared__ __align__(16) _Float16 atile[16 * 264];

  const int wave = threadIdx.x >> 5, lane = threadIdx.x & 31;
  const int hi = lane >> 4, l16 = lane & 15;
  const int strip = blockIdx.x;                          // 0..2047
  const unsigned t = threadIdx.x;
  const unsigned ldsA = lds_addr_of(&atile[0]);

#pragma unroll
  for (int i = 0; i < 4; ++i) {                          // 512 granules
    const unsigned g = t + i * 128;
    const int row = g >> 5, seg = g & 31;
    async_copy16(ldsA + row * 528 + seg * 16,
                 xh + (size_t)(strip * 16 + row) * CC + seg * 8);
  }
  wait_async0();
  __syncthreads();

  const _Float16* arow = &atile[l16 * 264];
  const _Float16* wr[4];
  v8f acc[4] = {};
#pragma unroll
  for (int u = 0; u < 4; ++u)
    wr[u] = qw + (size_t)((wave * 4 + u) * 16 + l16) * CC;

#pragma unroll
  for (int k0 = 0; k0 < CC; k0 += 32) {
    v16h af = frag_a_row(arow, k0, hi);
#pragma unroll
    for (int u = 0; u < 4; ++u)
      acc[u] = wmma_f16(af, frag_b_col(wr[u], k0, hi), acc[u]);
  }

#pragma unroll
  for (int u = 0; u < 4; ++u) {
    const int n = (wave * 4 + u) * 16 + l16;
    const float bias = q_b[n];
    const int h = n >> 5, dd = n & 31;
#pragma unroll
    for (int r = 0; r < 8; ++r) {
      int row = strip * 16 + hi * 8 + r;                 // b*4096+nq
      int b_ = row >> 12, nq = row & 4095;
      qbuf[(((size_t)(b_ * NHEAD + h) * NQ) + nq) * DHEAD + dd] =
          (_Float16)(acc[u][r] + bias);
    }
  }
}

// ---------------------------------------------------------------------------
// Stage 5: attention. Block = 8 waves = 128 queries of one (b,h).
// K (padded 48 halfs/row) and V^T (264 halfs/row) async-staged once to LDS.
// QK^T: one WMMA per 16x16 score tile; softmax in regs; probs through padded
// LDS for the C->A transpose; P*V from LDS.
// ---------------------------------------------------------------------------
__global__ __launch_bounds__(256)
void attn_kernel(const _Float16* __restrict__ qbuf,      // (64,4096,32)
                 const _Float16* __restrict__ kbuf,      // (64,256,32)
                 const _Float16* __restrict__ vtb,       // (64,32,256)
                 _Float16* __restrict__ ctx) {           // (B,4096,256)
  __shared__ __align__(32) _Float16 ldsK[256 * 48];      // 24 KB, 96B rows
  __shared__ __align__(16) _Float16 ldsV[32 * 264];      // 16.5 KB
  __shared__ __align__(16) _Float16 lds_p[8][16][264];   // 66 KB

  const int wave = threadIdx.x >> 5, lane = threadIdx.x & 31;
  const int hi = lane >> 4, l16 = lane & 15;
  const int bh = blockIdx.x >> 5;                        // 0..63
  const int qblk = blockIdx.x & 31;
  const int q0 = (qblk * 8 + wave) * 16;
  const unsigned t = threadIdx.x;

  // ---- stage K and V^T (1024 granules each) ----
  const _Float16* kg = kbuf + (size_t)bh * MRED * DHEAD;
  const _Float16* vg = vtb + (size_t)bh * DHEAD * MRED;
  const unsigned ldsKb = lds_addr_of(&ldsK[0]);
  const unsigned ldsVb = lds_addr_of(&ldsV[0]);
#pragma unroll
  for (int i = 0; i < 4; ++i) {
    const unsigned g = t + i * 256;
    {
      const int row = g >> 2, seg = g & 3;               // K: 256 rows x 4
      async_copy16(ldsKb + row * 96 + seg * 16, kg + row * 32 + seg * 8);
    }
    {
      const int row = g >> 5, seg = g & 31;              // V^T: 32 rows x 32
      async_copy16(ldsVb + row * 528 + seg * 16, vg + row * 256 + seg * 8);
    }
  }
  wait_async0();
  __syncthreads();

  // ---- Q fragment (whole head dim = one K=32 chunk) ----
  const _Float16* qbase = qbuf + ((size_t)bh * NQ + q0) * DHEAD;
  v16h qa = frag_a_row(qbase + (size_t)l16 * DHEAD, 0, hi);

  // ---- scores: 16 tiles of 16 keys ----
  v8f sc[16];
#pragma unroll
  for (int nt = 0; nt < 16; ++nt) {
    v16h bf = frag_b_col(&ldsK[(nt * 16 + l16) * 48], 0, hi);
    v8f z = {};
    sc[nt] = wmma_f16(qa, bf, z);
  }

  const float scale = 0.17677669529663687f;              // 1/sqrt(32)
#pragma unroll
  for (int nt = 0; nt < 16; ++nt)
#pragma unroll
    for (int r = 0; r < 8; ++r) sc[nt][r] *= scale;

  // ---- row softmax: columns live across each 16-lane half ----
#pragma unroll
  for (int r = 0; r < 8; ++r) {
    float mx = sc[0][r];
#pragma unroll
    for (int nt = 1; nt < 16; ++nt) mx = fmaxf(mx, sc[nt][r]);
#pragma unroll
    for (int off = 8; off > 0; off >>= 1) mx = fmaxf(mx, __shfl_xor(mx, off, 32));
    float sum = 0.f;
#pragma unroll
    for (int nt = 0; nt < 16; ++nt) {
      float pv = __expf(sc[nt][r] - mx);
      sc[nt][r] = pv;
      sum += pv;
    }
#pragma unroll
    for (int off = 8; off > 0; off >>= 1) sum += __shfl_xor(sum, off, 32);
    const float inv = 1.f / sum;
#pragma unroll
    for (int nt = 0; nt < 16; ++nt) sc[nt][r] *= inv;
  }

  // ---- transpose probs through LDS (C-layout -> row-major) ----
  _Float16(*P)[264] = lds_p[wave];
#pragma unroll
  for (int nt = 0; nt < 16; ++nt)
#pragma unroll
    for (int r = 0; r < 8; ++r)
      P[hi * 8 + r][nt * 16 + l16] = (_Float16)sc[nt][r];
  __syncthreads();

  // ---- P(16x256) * V(256x32) ----
  v8f o0 = {}, o1 = {};
#pragma unroll
  for (int kc = 0; kc < 8; ++kc) {
    const int k0 = kc * 32;
    v16h af = frag_a_row(&P[l16][0], k0, hi);
    v16h b0 = frag_b_col(&ldsV[(l16) * 264], k0, hi);      // d 0..15
    v16h b1 = frag_b_col(&ldsV[(16 + l16) * 264], k0, hi); // d 16..31
    o0 = wmma_f16(af, b0, o0);
    o1 = wmma_f16(af, b1, o1);
  }

  const int b_ = bh >> 3, h = bh & 7;
#pragma unroll
  for (int r = 0; r < 8; ++r) {
    int q = q0 + hi * 8 + r;
    size_t base = ((size_t)b_ * NQ + q) * CC + h * DHEAD;
    ctx[base + l16]      = (_Float16)o0[r];
    ctx[base + 16 + l16] = (_Float16)o1[r];
  }
}

// ---------------------------------------------------------------------------
// Stage 6: output projection (32768 x 256, K=256) -> fp32 d_out.
// Same structure as q_gemm.
// ---------------------------------------------------------------------------
__global__ __launch_bounds__(128)
void out_gemm_kernel(const _Float16* __restrict__ ctx,   // (32768,256)
                     const _Float16* __restrict__ ow,    // (256,256)
                     const float* __restrict__ out_b,
                     float* __restrict__ out) {          // (32768,256) f32
  __shared__ __align__(16) _Float16 atile[16 * 264];

  const int wave = threadIdx.x >> 5, lane = threadIdx.x & 31;
  const int hi = lane >> 4, l16 = lane & 15;
  const int strip = blockIdx.x;
  const unsigned t = threadIdx.x;
  const unsigned ldsA = lds_addr_of(&atile[0]);

#pragma unroll
  for (int i = 0; i < 4; ++i) {
    const unsigned g = t + i * 128;
    const int row = g >> 5, seg = g & 31;
    async_copy16(ldsA + row * 528 + seg * 16,
                 ctx + (size_t)(strip * 16 + row) * CC + seg * 8);
  }
  wait_async0();
  __syncthreads();

  const _Float16* arow = &atile[l16 * 264];
  const _Float16* wr[4];
  v8f acc[4] = {};
#pragma unroll
  for (int u = 0; u < 4; ++u)
    wr[u] = ow + (size_t)((wave * 4 + u) * 16 + l16) * CC;

#pragma unroll
  for (int k0 = 0; k0 < CC; k0 += 32) {
    v16h af = frag_a_row(arow, k0, hi);
#pragma unroll
    for (int u = 0; u < 4; ++u)
      acc[u] = wmma_f16(af, frag_b_col(wr[u], k0, hi), acc[u]);
  }

#pragma unroll
  for (int u = 0; u < 4; ++u) {
    const int n = (wave * 4 + u) * 16 + l16;
    const float bias = out_b[n];
#pragma unroll
    for (int r = 0; r < 8; ++r) {
      int row = strip * 16 + hi * 8 + r;
      out[(size_t)row * CC + n] = acc[u][r] + bias;
    }
  }
}

// ---------------------------------------------------------------------------
extern "C" void kernel_launch(void* const* d_in, const int* in_sizes, int n_in,
                              void* d_out, int out_size, void* d_ws,
                              size_t ws_size, hipStream_t stream) {
  const float* x      = (const float*)d_in[0];
  const float* conv_w = (const float*)d_in[1];
  const float* conv_b = (const float*)d_in[2];
  const float* ln_g   = (const float*)d_in[3];
  const float* ln_b   = (const float*)d_in[4];
  const float* kv_w   = (const float*)d_in[5];
  const float* kv_b   = (const float*)d_in[6];
  const float* q_w    = (const float*)d_in[7];
  const float* q_b    = (const float*)d_in[8];
  const float* out_w  = (const float*)d_in[9];
  const float* out_b  = (const float*)d_in[10];
  float* out = (float*)d_out;

  char* p = (char*)d_ws;
  auto carve = [&](size_t bytes) -> void* {
    void* r = (void*)p;
    p += (bytes + 255) & ~(size_t)255;
    return r;
  };
  _Float16* xh   = (_Float16*)carve((size_t)BZ * NQ * CC * 2);        // 16 MB
  _Float16* wc   = (_Float16*)carve((size_t)CC * KCONV * 2);          // 2 MB
  _Float16* kvwh = (_Float16*)carve((size_t)512 * CC * 2);
  _Float16* qwh  = (_Float16*)carve((size_t)CC * CC * 2);
  _Float16* owh  = (_Float16*)carve((size_t)CC * CC * 2);
  float*    x1p  = (float*)carve((size_t)BZ * MRED * CC * 4);         // 2 MB
  _Float16* x1h  = (_Float16*)carve((size_t)BZ * MRED * CC * 2);
  _Float16* kbuf = (_Float16*)carve((size_t)BZ * NHEAD * MRED * DHEAD * 2);
  _Float16* vtb  = (_Float16*)carve((size_t)BZ * NHEAD * DHEAD * MRED * 2);
  _Float16* qbuf = (_Float16*)carve((size_t)BZ * NHEAD * NQ * DHEAD * 2); // 16 MB
  _Float16* ctx  = (_Float16*)carve((size_t)BZ * NQ * CC * 2);            // 16 MB

  f32_to_f16_kernel<<<2048, 256, 0, stream>>>(x, xh, (size_t)BZ * NQ * CC);
  f32_to_f16_kernel<<<128, 256, 0, stream>>>(kv_w, kvwh, (size_t)512 * CC);
  f32_to_f16_kernel<<<64, 256, 0, stream>>>(q_w, qwh, (size_t)CC * CC);
  f32_to_f16_kernel<<<64, 256, 0, stream>>>(out_w, owh, (size_t)CC * CC);
  pack_conv_w_kernel<<<4096, 256, 0, stream>>>(conv_w, wc);

  conv_gemm_kernel<<<128, 256, 0, stream>>>(xh, wc, conv_b, x1p);
  ln_kernel<<<256, 256, 0, stream>>>(x1p, ln_g, ln_b, x1h);
  kv_gemm_kernel<<<128, 256, 0, stream>>>(x1h, kvwh, kv_b, kbuf, vtb);
  q_gemm_kernel<<<2048, 128, 0, stream>>>(xh, qwh, q_b, qbuf);
  attn_kernel<<<2048, 256, 0, stream>>>(qbuf, kbuf, vtb, ctx);
  out_gemm_kernel<<<2048, 128, 0, stream>>>(ctx, owh, out_b, out);
}
